// LSTM_15066745274638
// MI455X (gfx1250) — matrix operand order repaired and checked by reading, not compiled, not run on hardware
//
#include <hip/hip_runtime.h>
#include <hip/hip_bf16.h>

// ---------------------------------------------------------------------------
// LSTM (B=4096, T=64, I=2, H=512) -> h at last timestep, [B, H] f32.
// Recurrent GEMM via v_wmma_f32_16x16x32_bf16 (CDNA5 / gfx1250, wave32).
//
// Roofline: 2*B*4H*H*T ~= 550 GFLOP dominated by h @ W_hh^T. W_hh (2 MB bf16)
// stays resident in the 192 MB L2; HBM traffic is negligible, so the kernel is
// tensor-core bound. Strategy: persistent kernel, one WG per 16-row batch
// tile, the sequential t-loop inside; W is re-streamed from L2 every step
// (cheap) while h/c state stays on-WGP (LDS / registers).
// ---------------------------------------------------------------------------

typedef __bf16 v16bf __attribute__((ext_vector_type(16)));
typedef __bf16 v8bf  __attribute__((ext_vector_type(8)));
typedef float  v8f   __attribute__((ext_vector_type(8)));

#define B_TOT   4096
#define T_LEN   64
#define H_DIM   512
#define G_DIM   2048          // 4*H
#define KT      16            // K tiles: 512 / 32
#define NT_ALL  128           // N tiles: 2048 / 16
#define HPAD    8             // bf16 elems of row padding (16B) -> bank spread
#define HSTRIDE (H_DIM + HPAD)

__device__ __forceinline__ float sigmoidf_fast(float x) {
    return 1.0f / (1.0f + __expf(-x));
}
__device__ __forceinline__ float tanhf_fast(float x) {
    float e = __expf(-2.0f * x);
    return (1.0f - e) / (1.0f + e);
}

// ---------------------------------------------------------------------------
// Repack W_hh [2048, 512] f32 (row-major, W_hh[g,k]) into bf16 B-fragment
// blocks. Block bi = nt*KT + kt holds the 32x16 (KxN) tile of W_hh^T as
// 32 lanes x 16 bf16 contiguous (one 32B load per lane in the main loop).
// Lane l: N = nt*16 + (l&15); K = kt*32 + (j>=8 ? 16 : 0) + (l>=16 ? 8 : 0) + (j&7), j=0..15.
// ---------------------------------------------------------------------------
__global__ void lstm_prep_w(const float* __restrict__ W_hh,
                            __bf16* __restrict__ Wp) {
    int idx = blockIdx.x * blockDim.x + threadIdx.x;  // 0 .. 128*16*32*16-1
    if (idx >= NT_ALL * KT * 32 * 16) return;
    int j  = idx & 15;
    int l  = (idx >> 4) & 31;
    int kt = (idx >> 9) & 15;
    int nt = idx >> 13;                               // 0..127
    int n_global = nt * 16 + (l & 15);                // gate dim (0..2047)
    int k_global = kt * 32 + ((j >> 3) << 4) + ((l >> 4) << 3) + (j & 7);     // h dim (0..511)
    Wp[idx] = (__bf16)W_hh[n_global * H_DIM + k_global];
}

// ---------------------------------------------------------------------------
// Persistent recurrent kernel: one WG per 16-row batch tile, 8 wave32 waves.
// Wave w owns columns [64w, 64w+64) of EVERY gate, so i/f/g/o recombine
// wave-locally in WMMA C-layout registers. h is double-buffered in LDS (bf16)
// -> one barrier per timestep.
// ---------------------------------------------------------------------------
__launch_bounds__(256, 1)
__global__ void lstm_core(const float* __restrict__ x,     // [B, T, 2]
                          const float* __restrict__ W_ih,  // [2048, 2]
                          const float* __restrict__ b_ih,  // [2048]
                          const float* __restrict__ b_hh,  // [2048]
                          const __bf16* __restrict__ Wp,   // repacked W_hh
                          float* __restrict__ out) {       // [B, 512]
    __shared__ __bf16 h_sh[2][16 * HSTRIDE];
    __shared__ float  x_sh[2][16][2];
    __shared__ float  wih_sh[G_DIM][2];
    __shared__ float  bsum_sh[G_DIM];

    const int tid  = threadIdx.x;
    const int lane = tid & 31;
    const int wave = tid >> 5;
    const int lhi  = lane >> 4;   // 0: rows 0..7 / K-low, 1: rows 8..15 / K-high
    const int ll   = lane & 15;
    const int wg   = blockIdx.x;  // batch tile: rows [wg*16, wg*16+16)

    // One-time LDS fills: fused bias and input weights; zero initial h.
    for (int n = tid; n < G_DIM; n += 256) {
        bsum_sh[n]   = b_ih[n] + b_hh[n];
        wih_sh[n][0] = W_ih[n * 2 + 0];
        wih_sh[n][1] = W_ih[n * 2 + 1];
    }
    for (int i = tid; i < 16 * HSTRIDE; i += 256)
        h_sh[0][i] = (__bf16)0.0f;
    if (tid < 16) {  // stage x[:, 0, :]
        int row = wg * 16 + tid;
        x_sh[0][tid][0] = x[(row * T_LEN + 0) * 2 + 0];
        x_sh[0][tid][1] = x[(row * T_LEN + 0) * 2 + 1];
    }

    // Cell state for this wave's 64 h-columns: 4 tiles of v8f (C layout).
    v8f c[4];
    #pragma unroll
    for (int j = 0; j < 4; ++j)
        c[j] = (v8f){0.f, 0.f, 0.f, 0.f, 0.f, 0.f, 0.f, 0.f};

    __syncthreads();

    for (int t = 0; t < T_LEN; ++t) {
        const int cur = t & 1;
        const int nxt = cur ^ 1;
        const int last = (t == T_LEN - 1);

        // Anti-LICM: launder an integer offset (not the pointer!) through an
        // opaque asm. The B-fragment addresses then depend on a per-iteration
        // unknown, so they cannot be hoisted out of the time loop (which
        // previously spilled 2K VGPRs of weights), while the base pointer
        // keeps its kernel-arg provenance -> address-space inference still
        // emits global_load_b128 (not flat_load, which would tie LOADcnt and
        // DScnt together and serialize against the LDS A-fragment loads).
        int toff = 0;
        asm volatile("" : "+s"(toff));
        const __bf16* Wt = Wp + toff;

        // Build all 16 A-fragments (16x32 bf16) from h_sh[cur].
        // 16-bit A layout: lane<16 holds row ll, K 0..7 then 16..23;
        //                  lane>=16 holds row ll, K 8..15 then 24..31.
        v16bf a[KT];
        const __bf16* hrow = &h_sh[cur][ll * HSTRIDE];
        #pragma unroll
        for (int kt = 0; kt < KT; ++kt) {
            const __bf16* p = hrow + kt * 32 + lhi * 8;
            v8bf lo = *(const v8bf*)(p);
            v8bf hi = *(const v8bf*)(p + 16);
            a[kt] = __builtin_shufflevector(lo, hi,
                0, 1, 2, 3, 4, 5, 6, 7, 8, 9, 10, 11, 12, 13, 14, 15);
        }

        // 4 column-slices of 16; each slice needs all four gates.
        #pragma unroll
        for (int j = 0; j < 4; ++j) {
            v8f acc[4];  // i, f, g, o accumulators for this slice
            #pragma unroll
            for (int q = 0; q < 4; ++q) {
                const int nt   = q * 32 + wave * 4 + j;  // N tile (0..127)
                const int ncol = nt * 16 + ll;           // gate column
                // Fused init: bias + x @ W_ih (I=2).
                const float w0 = wih_sh[ncol][0];
                const float w1 = wih_sh[ncol][1];
                const float bs = bsum_sh[ncol];
                v8f av;
                #pragma unroll
                for (int r = 0; r < 8; ++r) {
                    int row = r + lhi * 8;
                    av[r] = bs + x_sh[cur][row][0] * w0 + x_sh[cur][row][1] * w1;
                }
                // K loop: 16 WMMAs, B streamed from L2-resident packed W_hh.
                const __bf16* wp = Wt + (size_t)(nt * KT) * 512 + lane * 16;
                #pragma unroll
                for (int kt = 0; kt < KT; ++kt) {
                    v16bf b = *(const v16bf*)(wp + kt * 512);
                    av = __builtin_amdgcn_wmma_f32_16x16x32_bf16(
                        false, a[kt], false, b, (short)0, av, false, false);
                }
                acc[q] = av;
            }
            // Pointwise LSTM update, wave-local (C layout: lane ll -> col,
            // vgpr r -> row r + lhi*8). Write next-step h to the other buffer.
            const int col = wave * 64 + j * 16 + ll;
            #pragma unroll
            for (int r = 0; r < 8; ++r) {
                float iv = sigmoidf_fast(acc[0][r]);
                float fv = sigmoidf_fast(acc[1][r]);
                float gv = tanhf_fast(acc[2][r]);
                float ov = sigmoidf_fast(acc[3][r]);
                float cv = fv * c[j][r] + iv * gv;
                c[j][r] = cv;
                float hv = ov * tanhf_fast(cv);
                int row = r + lhi * 8;
                if (last) {
                    out[(size_t)(wg * 16 + row) * H_DIM + col] = hv;
                } else {
                    h_sh[nxt][row * HSTRIDE + col] = (__bf16)hv;
                }
            }
            // Scheduling fence: keep the load/WMMA window to one slice so the
            // scheduler doesn't cluster 256 loads ahead and re-spill.
            asm volatile("" ::: "memory");
        }

        // Stage x for the next timestep into the other buffer.
        if (!last && tid < 16) {
            int row = wg * 16 + tid;
            x_sh[nxt][tid][0] = x[(row * T_LEN + (t + 1)) * 2 + 0];
            x_sh[nxt][tid][1] = x[(row * T_LEN + (t + 1)) * 2 + 1];
        }
        __syncthreads();  // h_sh[nxt]/x_sh[nxt] complete before next step
    }
}

extern "C" void kernel_launch(void* const* d_in, const int* in_sizes, int n_in,
                              void* d_out, int out_size, void* d_ws, size_t ws_size,
                              hipStream_t stream) {
    const float* x    = (const float*)d_in[0];
    const float* W_ih = (const float*)d_in[1];
    const float* W_hh = (const float*)d_in[2];
    const float* b_ih = (const float*)d_in[3];
    const float* b_hh = (const float*)d_in[4];
    __bf16* Wp  = (__bf16*)d_ws;          // 2 MB packed bf16 weights
    float*  out = (float*)d_out;

    // Repack W_hh -> bf16 B-fragment blocks (1M elements).
    lstm_prep_w<<<4096, 256, 0, stream>>>(W_hh, Wp);
    // Persistent recurrent kernel: 256 batch tiles x 256 threads (8 waves).
    lstm_core<<<B_TOT / 16, 256, 0, stream>>>(x, W_ih, b_ih, b_hh, Wp, out);
}
